// ImprovedFragranceGNN_46755013984596
// MI455X (gfx1250) — matrix operand order repaired and checked by reading, not compiled, not run on hardware
//
#include <hip/hip_runtime.h>
#include <hip/hip_bf16.h>

// Problem constants (match reference)
#define NN   100000
#define EE   800000
#define GG   1024
#define DIN  128
#define HH   256
#define LL   256
#define EPSV 1e-5f

typedef unsigned short u16;
typedef unsigned int   u32;

typedef __attribute__((ext_vector_type(16))) __bf16 v16bf;
typedef __attribute__((ext_vector_type(8)))  float  v8f;

__device__ __forceinline__ u16 f2b(float f) {
  u32 u = __builtin_bit_cast(u32, f);
  u32 r = (u + 0x7FFFu + ((u >> 16) & 1u)) >> 16;   // round-nearest-even
  return (u16)r;
}
__device__ __forceinline__ float b2f(u16 s) {
  u32 u = ((u32)s) << 16;
  return __builtin_bit_cast(float, u);
}

// ---------------------------------------------------------------------------
// Generic bf16 WMMA GEMM:  out = act(A[M,K](bf16) @ Wt^T + bias)
// Wt is stored transposed: Wt[n*K + k] = W[k][n] (bf16).
// Block = 128 threads = 4 waves (2x2). Wave tile = 16 rows x 64 cols.
// Grid = (M/32, Nout/128). M % 32 == 0, Nout % 128 == 0 guaranteed by caller.
// ---------------------------------------------------------------------------
__launch_bounds__(128)
__global__ void k_gemm_bf16(const u16* __restrict__ A, int lda,
                            const u16* __restrict__ Wt, int K,
                            const float* __restrict__ bias,
                            float* __restrict__ outF, int ldoF,
                            u16* __restrict__ outB, int ldoB, int colOff,
                            int relu)
{
  const int lane = threadIdx.x & 31;
  const int wave = threadIdx.x >> 5;
  const int half = lane >> 4;       // 0: lanes 0-15, 1: lanes 16-31
  const int lm   = lane & 15;
  const int row0 = blockIdx.x * 32 + (wave & 1) * 16;
  const int col0 = blockIdx.y * 128 + (wave >> 1) * 64;

  const u16* arow = A + (size_t)(row0 + lm) * lda;

  v8f acc[4] = {};   // 4 N-tiles of 16x16 f32 accumulators

  for (int kb = 0; kb < K; kb += 32) {
    // A fragment: 16x32 bf16, ISA layout: VGPR v -> k = (v/4)*16 + half*8 + (v%4)*2
    union { v16bf v; u32 u[8]; } a;
    #pragma unroll
    for (int v = 0; v < 8; ++v) {
      int k = kb + ((v >> 2) << 4) + (half << 3) + ((v & 3) << 1);
      a.u[v] = *(const u32*)(arow + k);
    }
    #pragma unroll
    for (int t = 0; t < 4; ++t) {
      // B fragment: 32x16 bf16, lane=col n; VGPR v -> k = (v/4)*16 + half*8 + (v%4)*2
      const u16* wrow = Wt + (size_t)(col0 + t * 16 + lm) * K;
      union { v16bf v; u32 u[8]; } b;
      #pragma unroll
      for (int v = 0; v < 8; ++v) {
        int k = kb + ((v >> 2) << 4) + (half << 3) + ((v & 3) << 1);
        b.u[v] = *(const u32*)(wrow + k);
      }
      acc[t] = __builtin_amdgcn_wmma_f32_16x16x32_bf16(
          false, a.v, false, b.v, (short)0, acc[t], false, false);
    }
  }

  // C/D layout: VGPR r, lanes 0-15 -> M=r, lanes 16-31 -> M=8+r, N=lane%16
  #pragma unroll
  for (int t = 0; t < 4; ++t) {
    int n = col0 + t * 16 + lm;
    float bv = bias ? bias[n] : 0.0f;
    #pragma unroll
    for (int r = 0; r < 8; ++r) {
      int m = row0 + (half << 3) + r;
      float val = acc[t][r] + bv;
      if (relu) val = fmaxf(val, 0.0f);
      if (outF) outF[(size_t)m * ldoF + n] = val;
      if (outB) outB[(size_t)m * ldoB + colOff + n] = f2b(val);
    }
  }
}

// ---------------------------------------------------------------------------
// Small utility / graph kernels
// ---------------------------------------------------------------------------
__global__ void k_zero_f32(float* __restrict__ p, int n) {
  int i = blockIdx.x * blockDim.x + threadIdx.x;
  if (i < n) p[i] = 0.0f;
}

__global__ void k_convert_bf16(u16* __restrict__ o, const float* __restrict__ x, int n) {
  int i = blockIdx.x * blockDim.x + threadIdx.x;
  if (i < n) o[i] = f2b(x[i]);
}

// Wt[n*K + k] = bf16(W[k*Nw + n])
__global__ void k_wt(const float* __restrict__ W, u16* __restrict__ Wt, int K, int Nw) {
  int i = blockIdx.x * blockDim.x + threadIdx.x;
  if (i >= K * Nw) return;
  int n = i / K, k = i - n * K;
  Wt[i] = f2b(W[(size_t)k * Nw + n]);
}

__global__ void k_deg(const int* __restrict__ dst, float* __restrict__ deg, int e) {
  int i = blockIdx.x * blockDim.x + threadIdx.x;
  if (i < e) atomicAdd(&deg[dst[i]], 1.0f);
}

__global__ void k_dinv(float* __restrict__ d, int n) {
  int i = blockIdx.x * blockDim.x + threadIdx.x;
  if (i < n) d[i] = __frsqrt_rn(1.0f + d[i]);
}

// agg[n,f] = xw[n,f]*dinv[n]^2 + bias[f]
__global__ void k_agg_init(float* __restrict__ agg, const float* __restrict__ xw,
                           const float* __restrict__ dinv, const float* __restrict__ bias,
                           int n, int F) {
  int i = blockIdx.x * blockDim.x + threadIdx.x;
  if (i >= n * F) return;
  int node = i / F, f = i - node * F;
  float dv = dinv[node];
  agg[i] = xw[i] * dv * dv + bias[f];
}

// wave per edge: agg[dst] += xw[src] * dinv[src]*dinv[dst]
__launch_bounds__(256)
__global__ void k_edge(const int* __restrict__ src, const int* __restrict__ dst,
                       const float* __restrict__ xw, const float* __restrict__ dinv,
                       float* __restrict__ agg, int e, int F) {
  int w = (blockIdx.x * 256 + threadIdx.x) >> 5;
  int lane = threadIdx.x & 31;
  if (w >= e) return;
  int s = src[w], d = dst[w];
  float coef = dinv[s] * dinv[d];
  const float4* xs = (const float4*)(xw + (size_t)s * F);
  float* ad = agg + (size_t)d * F;
  for (int j = lane; j < (F >> 2); j += 32) {
    float4 v = xs[j];
    int f = j << 2;
    atomicAdd(&ad[f + 0], v.x * coef);
    atomicAdd(&ad[f + 1], v.y * coef);
    atomicAdd(&ad[f + 2], v.z * coef);
    atomicAdd(&ad[f + 3], v.w * coef);
  }
}

__global__ void k_relu_bf16(u16* __restrict__ o, const float* __restrict__ x, int n) {
  int i = blockIdx.x * blockDim.x + threadIdx.x;
  if (i < n) o[i] = f2b(fmaxf(x[i], 0.0f));
}

// wave per node: mean-pool accumulation over 128 features (bf16 input)
__launch_bounds__(256)
__global__ void k_pool(const u16* __restrict__ h, const int* __restrict__ batch,
                       float* __restrict__ sums, float* __restrict__ cnt, int n) {
  int w = (blockIdx.x * 256 + threadIdx.x) >> 5;
  int lane = threadIdx.x & 31;
  if (w >= n) return;
  int g = batch[w];
  const u16* hr = h + (size_t)w * 128;
  float* sg = sums + (size_t)g * 128;
  #pragma unroll
  for (int j = 0; j < 4; ++j) {
    int f = lane * 4 + j;
    atomicAdd(&sg[f], b2f(hr[f]));
  }
  if (lane == 0) atomicAdd(&cnt[g], 1.0f);
}

__global__ void k_pool_final(u16* __restrict__ combined, const float* __restrict__ sums,
                             const float* __restrict__ cnt, int colOff) {
  int i = blockIdx.x * blockDim.x + threadIdx.x;
  if (i >= GG * 128) return;
  int g = i >> 7, f = i & 127;
  float c = fmaxf(cnt[g], 1.0f);
  combined[(size_t)g * 512 + colOff + f] = f2b(sums[i] / c);
}

// out_bf = bn(x) = g*(x-m)*rsqrt(v+eps)+b  (input already ReLU'd by GEMM epilogue)
__global__ void k_bn(u16* __restrict__ o, const float* __restrict__ x,
                     const float* __restrict__ gg, const float* __restrict__ bb,
                     const float* __restrict__ mm, const float* __restrict__ vv,
                     int n, int width) {
  int i = blockIdx.x * blockDim.x + threadIdx.x;
  if (i >= n) return;
  int f = i % width;
  float y = gg[f] * (x[i] - mm[f]) * __frsqrt_rn(vv[f] + EPSV) + bb[f];
  o[i] = f2b(y);
}

// ---------------------------------------------------------------------------
static inline int cdiv(int a, int b) { return (a + b - 1) / b; }

extern "C" void kernel_launch(void* const* d_in, const int* in_sizes, int n_in,
                              void* d_out, int out_size, void* d_ws, size_t ws_size,
                              hipStream_t stream) {
  (void)in_sizes; (void)n_in; (void)out_size; (void)ws_size;

  const float* x_in[2]   = {(const float*)d_in[0], (const float*)d_in[4]};
  const int*   edge[2]   = {(const int*)d_in[1],   (const int*)d_in[5]};
  const int*   batch[2]  = {(const int*)d_in[2],   (const int*)d_in[6]};
  const float* notes[2]  = {(const float*)d_in[3], (const float*)d_in[7]};

  // params in dict insertion order
  const float* W[9]  = {(const float*)d_in[8],  (const float*)d_in[10], (const float*)d_in[12],
                        (const float*)d_in[14], (const float*)d_in[16], (const float*)d_in[18],
                        (const float*)d_in[20], (const float*)d_in[22], (const float*)d_in[24]};
  const int    WK[9] = {128, 256, 256, 256, 256, 256, 512, 256, 128};
  const int    WN[9] = {256, 256, 256, 128, 256, 128, 256, 128, 256};
  const float* B[9]  = {(const float*)d_in[9],  (const float*)d_in[11], (const float*)d_in[13],
                        (const float*)d_in[15], (const float*)d_in[17], (const float*)d_in[19],
                        (const float*)d_in[21], (const float*)d_in[23], (const float*)d_in[25]};
  const float* bn1g = (const float*)d_in[26]; const float* bn1b = (const float*)d_in[27];
  const float* bn1m = (const float*)d_in[28]; const float* bn1v = (const float*)d_in[29];
  const float* bn2g = (const float*)d_in[30]; const float* bn2b = (const float*)d_in[31];
  const float* bn2m = (const float*)d_in[32]; const float* bn2v = (const float*)d_in[33];

  float* out = (float*)d_out;

  // ---- workspace carve-out ----
  char* ws = (char*)d_ws;
  size_t off = 0;
  auto alloc = [&](size_t bytes) -> char* {
    char* p = ws + off;
    off += (bytes + 255) & ~(size_t)255;
    return p;
  };
  float* dinv     = (float*)alloc((size_t)NN * 4);
  u16*   xbf      = (u16*)  alloc((size_t)NN * 256 * 2);
  float* xw       = (float*)alloc((size_t)NN * 256 * 4);
  float* agg      = (float*)alloc((size_t)NN * 256 * 4);
  float* pooled   = (float*)alloc((size_t)GG * 128 * 4);
  float* cnt      = (float*)alloc((size_t)GG * 4);
  u16*   combined = (u16*)  alloc((size_t)GG * 512 * 2);
  float* headF    = (float*)alloc((size_t)GG * 256 * 4);
  u16*   headB    = (u16*)  alloc((size_t)GG * 256 * 2);
  u16*   nbuf     = (u16*)  alloc((size_t)GG * 256 * 2);
  u16*   nh       = (u16*)  alloc((size_t)GG * 256 * 2);
  u16*   wt[9];
  for (int i = 0; i < 9; ++i) wt[i] = (u16*)alloc((size_t)WK[i] * WN[i] * 2);

  // ---- weight prep: transpose + bf16 convert (cheap, ~1 MB total) ----
  for (int i = 0; i < 9; ++i) {
    int ne = WK[i] * WN[i];
    k_wt<<<cdiv(ne, 256), 256, 0, stream>>>(W[i], wt[i], WK[i], WN[i]);
  }

  // ---- two GNN branches ----
  const int layKin[4]  = {128, 256, 256, 256};
  const int layKout[4] = {256, 256, 256, 128};
  for (int b = 0; b < 2; ++b) {
    const int* src = edge[b];
    const int* dst = edge[b] + EE;

    k_zero_f32<<<cdiv(NN, 256), 256, 0, stream>>>(dinv, NN);
    k_deg<<<cdiv(EE, 256), 256, 0, stream>>>(dst, dinv, EE);
    k_dinv<<<cdiv(NN, 256), 256, 0, stream>>>(dinv, NN);

    k_convert_bf16<<<cdiv(NN * DIN, 256), 256, 0, stream>>>(xbf, x_in[b], NN * DIN);

    for (int l = 0; l < 4; ++l) {
      int kin = layKin[l], kout = layKout[l];
      dim3 gg(NN / 32, kout / 128);
      k_gemm_bf16<<<gg, 128, 0, stream>>>(xbf, kin, wt[l], kin, nullptr,
                                          xw, kout, nullptr, 0, 0, 0);
      k_agg_init<<<cdiv(NN * kout, 256), 256, 0, stream>>>(agg, xw, dinv, B[l], NN, kout);
      k_edge<<<cdiv(EE * 32, 256), 256, 0, stream>>>(src, dst, xw, dinv, agg, EE, kout);
      k_relu_bf16<<<cdiv(NN * kout, 256), 256, 0, stream>>>(xbf, agg, NN * kout);
    }

    k_zero_f32<<<cdiv(GG * 128, 256), 256, 0, stream>>>(pooled, GG * 128);
    k_zero_f32<<<cdiv(GG, 256), 256, 0, stream>>>(cnt, GG);
    k_pool<<<cdiv(NN * 32, 256), 256, 0, stream>>>(xbf, batch[b], pooled, cnt, NN);
    k_pool_final<<<cdiv(GG * 128, 256), 256, 0, stream>>>(combined, pooled, cnt, b * 128);
  }

  // ---- notes branches -> combined cols [256..512) ----
  for (int b = 0; b < 2; ++b) {
    k_convert_bf16<<<cdiv(GG * LL, 256), 256, 0, stream>>>(nbuf, notes[b], GG * LL);
    dim3 g1(GG / 32, 2);  // nW1: 256->256
    k_gemm_bf16<<<g1, 128, 0, stream>>>(nbuf, 256, wt[4], 256, B[4],
                                        nullptr, 0, nh, 256, 0, 1);
    dim3 g2(GG / 32, 1);  // nW2: 256->128, write into combined
    k_gemm_bf16<<<g2, 128, 0, stream>>>(nh, 256, wt[5], 256, B[5],
                                        nullptr, 0, combined, 512, 256 + b * 128, 1);
  }

  // ---- head ----
  {
    dim3 g1(GG / 32, 2);  // cW1: 512->256, ReLU
    k_gemm_bf16<<<g1, 128, 0, stream>>>(combined, 512, wt[6], 512, B[6],
                                        headF, 256, nullptr, 0, 0, 1);
    k_bn<<<cdiv(GG * 256, 256), 256, 0, stream>>>(headB, headF, bn1g, bn1b, bn1m, bn1v,
                                                  GG * 256, 256);
    dim3 g2(GG / 32, 1);  // cW2: 256->128, ReLU
    k_gemm_bf16<<<g2, 128, 0, stream>>>(headB, 256, wt[7], 256, B[7],
                                        headF, 128, nullptr, 0, 0, 1);
    k_bn<<<cdiv(GG * 128, 256), 256, 0, stream>>>(headB, headF, bn2g, bn2b, bn2m, bn2v,
                                                  GG * 128, 128);
    dim3 g3(GG / 32, 2);  // cW3: 128->256, no activation -> d_out
    k_gemm_bf16<<<g3, 128, 0, stream>>>(headB, 128, wt[8], 128, B[8],
                                        out, 256, nullptr, 0, 0, 0);
  }
}
